// MP_GE_50036368998502
// MI455X (gfx1250) — compile-verified, matrix operand order checked by reading, not run on hardware
//
#include <hip/hip_runtime.h>
#include <hip/hip_bf16.h>
#include <math.h>

// Problem constants (match reference)
#define BGR   64
#define NNODE 1000
#define KNN   10
#define LAY   4
#define EMB   128
#define NT    (BGR * NNODE)    // 64000 nodes
#define NE    (NT * KNN)       // 640000 edges
#define INF_  5

typedef __attribute__((ext_vector_type(16))) __bf16 v16bf;
typedef __attribute__((ext_vector_type(8)))  __bf16 v8bf;
typedef __attribute__((ext_vector_type(8)))  float  v8f;
typedef __attribute__((ext_vector_type(4)))  unsigned v4u;
typedef __attribute__((ext_vector_type(8)))  unsigned v8u;

// ---------------------------------------------------------------------------
// TDM: DMA a 2D bf16 tile (tile0 x tile1 elements, global row stride stride0
// elements) from global memory into LDS (dense tile0-element rows).
// 2-group descriptor form (tensors up to 2D), per CDNA5 ISA.
// Issued by one wave; caller must s_wait_tensorcnt + barrier.
// ---------------------------------------------------------------------------
__device__ __forceinline__ void tdm_load_2d_bf16(const void* gptr, unsigned lds_off,
                                                 unsigned tdim0, unsigned tdim1,
                                                 unsigned tile0, unsigned tile1,
                                                 unsigned stride0) {
    unsigned long long ga = (unsigned long long)gptr;
    v4u g0;
    g0[0] = 1u;                                   // count=1, user mode
    g0[1] = lds_off;                              // lds_addr (bytes)
    g0[2] = (unsigned)(ga & 0xffffffffu);         // global_addr[31:0]
    g0[3] = (unsigned)((ga >> 32) & 0x01ffffffu)  // global_addr[56:32]
          | (2u << 30);                           // type = 2 ("image")
    v8u g1;
    g1[0] = (1u << 16);                           // workgroup_mask=0, data_size=1 (2B)
    g1[1] = (tdim0 & 0xffffu) << 16;              // tensor_dim0 lo16 in [63:48]
    g1[2] = ((tdim0 >> 16) & 0xffffu)             // tensor_dim0 hi16
          | ((tdim1 & 0xffffu) << 16);            // tensor_dim1 lo16
    g1[3] = ((tdim1 >> 16) & 0xffffu)             // tensor_dim1 hi16
          | (tile0 << 16);                        // tile_dim0
    g1[4] = tile1;                                // tile_dim1 (tile_dim2 = 0)
    g1[5] = stride0;                              // tensor_dim0_stride[31:0]
    g1[6] = 0;                                    // stride0 hi, stride1 lo
    g1[7] = 0;                                    // stride1 hi
    asm volatile("tensor_load_to_lds %0, %1" :: "s"(g0), "s"(g1) : "memory");
}

// ---------------------------------------------------------------------------
// Weight prep: fp32 [L][K][N] -> bf16 transposed [L][N][K] (K contiguous per
// output column -> TDM-friendly rows; B-fragments become contiguous chunks).
// ---------------------------------------------------------------------------
__global__ void cvt_wmsg_kernel(const float* __restrict__ W, __bf16* __restrict__ WT) {
    int t = blockIdx.x * blockDim.x + threadIdx.x;
    const int per = 257 * EMB;
    if (t >= LAY * per) return;
    int l = t / per, rem = t - l * per;
    int k = rem / EMB, n = rem - k * EMB;
    WT[((size_t)l * EMB + n) * 257 + k] = (__bf16)W[t];
}
__global__ void cvt_wupd_kernel(const float* __restrict__ W, __bf16* __restrict__ WT) {
    int t = blockIdx.x * blockDim.x + threadIdx.x;
    const int per = 256 * EMB;
    if (t >= LAY * per) return;
    int l = t / per, rem = t - l * per;
    int k = rem / EMB, n = rem - k * EMB;
    WT[((size_t)l * EMB + n) * 256 + k] = (__bf16)W[t];
}

// ---------------------------------------------------------------------------
// Input embedding: hbf = bf16(relu(h_feat @ W_emb + b_emb)); copy pos to ws.
// All GEMM consumers of activations are bf16 WMMA, so activations live in bf16.
// ---------------------------------------------------------------------------
__global__ void embed_kernel(const float* __restrict__ hf, const float* __restrict__ We,
                             const float* __restrict__ be, __bf16* __restrict__ hbf,
                             const float* __restrict__ pos_in, float* __restrict__ pos) {
    int t = blockIdx.x * blockDim.x + threadIdx.x;
    if (t < NT * EMB) {
        int n = t >> 7, c = t & 127;
        float s = be[c];
#pragma unroll
        for (int f = 0; f < INF_; ++f) s += hf[n * INF_ + f] * We[f * EMB + c];
        hbf[t] = (__bf16)(s > 0.f ? s : 0.f);
    }
    if (t < NT * 2) pos[t] = pos_in[t];
}

__global__ void zero_kernel(float* __restrict__ p, int n) {
    int t = blockIdx.x * blockDim.x + threadIdx.x;
    if (t < n) p[t] = 0.f;
}

// ---------------------------------------------------------------------------
// Per-graph kNN (layers >= 1). Positions LDS-resident, register insertion sort.
// ---------------------------------------------------------------------------
__global__ void knn_kernel(const float* __restrict__ pos, int* __restrict__ src,
                           int* __restrict__ dst) {
    __shared__ float2 sp[NNODE];
    int g = blockIdx.y;
    for (int j = threadIdx.x; j < NNODE; j += blockDim.x)
        sp[j] = ((const float2*)pos)[g * NNODE + j];
    __syncthreads();
    int i = blockIdx.x * blockDim.x + threadIdx.x;
    if (i >= NNODE) return;
    float2 pi = sp[i];
    float bd[KNN]; int bi[KNN];
#pragma unroll
    for (int k = 0; k < KNN; ++k) { bd[k] = 3.0e38f; bi[k] = 0; }
    for (int j = 0; j < NNODE; ++j) {
        if (j == i) continue;
        float dx = pi.x - sp[j].x, dy = pi.y - sp[j].y;
        float d2 = dx * dx + dy * dy;
        if (d2 < bd[KNN - 1]) {
            bd[KNN - 1] = d2; bi[KNN - 1] = j;
#pragma unroll
            for (int p = KNN - 1; p > 0; --p) {
                if (bd[p] < bd[p - 1]) {
                    float td = bd[p]; bd[p] = bd[p - 1]; bd[p - 1] = td;
                    int   ti = bi[p]; bi[p] = bi[p - 1]; bi[p - 1] = ti;
                }
            }
        }
    }
    int node = g * NNODE + i;
#pragma unroll
    for (int k = 0; k < KNN; ++k) {
        int e = node * KNN + k;
        src[e] = g * NNODE + bi[k];
        dst[e] = node;
    }
}

__global__ void cnt_kernel(const int* __restrict__ dst, float* __restrict__ cnt) {
    int t = blockIdx.x * blockDim.x + threadIdx.x;
    if (t < NE) atomicAdd(&cnt[dst[t]], 1.0f);
}

// ---------------------------------------------------------------------------
// Edge message kernel: one wave = 16 edges x 128 outputs.
//   A[e, 0:128]=h[dst], A[e,128:256]=h[src] (bf16 gathers, no cvt),
//   A[e,256]=|pos diff| handled as rank-1 VALU update.
//   msg = relu(A @ Wm + bm); gate = relu(msg @ Wp + bp)
//   atomic scatter: msg -> msg_aggr[dst], pos_diff*gate -> pos_aggr[dst]
// Weights (first 256 K-rows) staged to LDS by TDM once per workgroup;
// B fragments come from ds_load_b128 (16B-aligned, 512B row stride).
// ---------------------------------------------------------------------------
__global__ __launch_bounds__(256)
void msg_kernel(const __bf16* __restrict__ hbf, const float* __restrict__ pos,
                const int* __restrict__ src, const int* __restrict__ dst,
                const __bf16* __restrict__ WmT,  // [128][257] global
                const float* __restrict__ bm, const float* __restrict__ Wp,
                const float* __restrict__ bp, float* __restrict__ msg_aggr,
                float* __restrict__ pos_aggr) {
    __shared__ __align__(16) __bf16 sW[EMB * 256];   // 64 KB: [col][k0..255]

    if (threadIdx.x < 32) {   // wave 0 drives the TDM (EXEC ignored by TDM)
        tdm_load_2d_bf16(WmT, (unsigned)(unsigned long long)(uintptr_t)sW,
                         257u, 128u, 256u, 128u, 257u);
        __builtin_amdgcn_s_wait_tensorcnt(0);
    }
    __syncthreads();

    int wave = blockIdx.x * (blockDim.x >> 5) + (threadIdx.x >> 5);
    int lane = threadIdx.x & 31;
    int half = lane >> 4;                       // K-half select (ISA A/B layout)
    int l15  = lane & 15;
    int e0   = wave * 16;

    // A-row gather pointers: lane covers matrix row M = l15
    int my_e = e0 + l15;
    const __bf16* hi = hbf + (size_t)dst[my_e] * EMB;   // features   0..127
    const __bf16* hj = hbf + (size_t)src[my_e] * EMB;   // features 128..255

    v8f C[8];
#pragma unroll
    for (int nt = 0; nt < 8; ++nt) C[nt] = (v8f){0, 0, 0, 0, 0, 0, 0, 0};

#pragma unroll
    for (int kc = 0; kc < 8; ++kc) {
        int kb = kc * 32 + half * 8;            // lane's two 8-elem K chunks
        const __bf16* f0 = (kb      < 128) ? (hi + kb)      : (hj + kb - 128);
        const __bf16* f1 = (kb + 16 < 128) ? (hi + kb + 16) : (hj + kb - 112);
        v8bf a0 = *(const v8bf*)f0;
        v8bf a1 = *(const v8bf*)f1;
        v16bf a = __builtin_shufflevector(a0, a1, 0, 1, 2, 3, 4, 5, 6, 7,
                                          8, 9, 10, 11, 12, 13, 14, 15);
#pragma unroll
        for (int nt = 0; nt < 8; ++nt) {
            const v8bf* wp = (const v8bf*)(sW + (((unsigned)(nt * 16 + l15)) << 8) + kb);
            v8bf b0 = wp[0];
            v8bf b1 = wp[2];                    // +16 elements
            v16bf b = __builtin_shufflevector(b0, b1, 0, 1, 2, 3, 4, 5, 6, 7,
                                              8, 9, 10, 11, 12, 13, 14, 15);
            C[nt] = __builtin_amdgcn_wmma_f32_16x16x32_bf16(
                false, a, false, b, (short)0, C[nt], false, false);
        }
    }

    // Distances / pos diffs for this lane-half's 8 rows (M = r + 8*half)
    float pdx[8], pdy[8], dl[8];
    int   dgt[8];
#pragma unroll
    for (int r = 0; r < 8; ++r) {
        int m = e0 + r + half * 8;
        int s = src[m], d = dst[m];
        dgt[r] = d;
        float dx = pos[(size_t)d * 2]     - pos[(size_t)s * 2];
        float dy = pos[(size_t)d * 2 + 1] - pos[(size_t)s * 2 + 1];
        pdx[r] = dx; pdy[r] = dy; dl[r] = sqrtf(dx * dx + dy * dy);
    }

    // Rank-1 dist column (K=256, from global) + bias + relu
#pragma unroll
    for (int nt = 0; nt < 8; ++nt) {
        int col = nt * 16 + l15;
        float wl = (float)WmT[(size_t)col * 257 + 256];
        float bi = bm[col];
#pragma unroll
        for (int r = 0; r < 8; ++r) {
            float v = C[nt][r] + dl[r] * wl + bi;
            C[nt][r] = v > 0.f ? v : 0.f;
        }
    }

    // gate = relu(msg @ Wp + bp); pos_aggr += pos_diff * gate (atomic)
#pragma unroll
    for (int r = 0; r < 8; ++r) {
        float p0 = 0.f, p1 = 0.f;
#pragma unroll
        for (int nt = 0; nt < 8; ++nt) {
            int col = nt * 16 + l15;
            float v = C[nt][r];
            p0 += v * Wp[col * 2 + 0];
            p1 += v * Wp[col * 2 + 1];
        }
#pragma unroll
        for (int m2 = 1; m2 < 16; m2 <<= 1) {   // 16-lane butterfly, wave32
            p0 += __shfl_xor(p0, m2, 32);
            p1 += __shfl_xor(p1, m2, 32);
        }
        if (l15 == 0) {
            float g0 = p0 + bp[0]; g0 = g0 > 0.f ? g0 : 0.f;
            float g1 = p1 + bp[1]; g1 = g1 > 0.f ? g1 : 0.f;
            atomicAdd(&pos_aggr[(size_t)dgt[r] * 2],     pdx[r] * g0);
            atomicAdd(&pos_aggr[(size_t)dgt[r] * 2 + 1], pdy[r] * g1);
        }
    }

    // Scatter-add messages
#pragma unroll
    for (int r = 0; r < 8; ++r) {
        int base = dgt[r] * EMB + l15;
#pragma unroll
        for (int nt = 0; nt < 8; ++nt)
            atomicAdd(&msg_aggr[base + nt * 16], C[nt][r]);
    }
}

// ---------------------------------------------------------------------------
// Update GEMM: htmp = relu([hbf | msg_aggr] @ Wu + bu). K 0..127 comes from
// bf16 activations (direct), K 128..255 from fp32 msg_aggr (cvt). Dense 64 KB
// weight tile staged to LDS by TDM.
// ---------------------------------------------------------------------------
__global__ __launch_bounds__(256)
void upd_kernel(const __bf16* __restrict__ hbf, const float* __restrict__ aggr,
                const __bf16* __restrict__ WuT,  // [128][256] global
                const float* __restrict__ bu, float* __restrict__ htmp) {
    __shared__ __align__(16) __bf16 sW[EMB * 256];   // 64 KB

    if (threadIdx.x < 32) {
        tdm_load_2d_bf16(WuT, (unsigned)(unsigned long long)(uintptr_t)sW,
                         256u, 128u, 256u, 128u, 256u);
        __builtin_amdgcn_s_wait_tensorcnt(0);
    }
    __syncthreads();

    int wave = blockIdx.x * (blockDim.x >> 5) + (threadIdx.x >> 5);
    int lane = threadIdx.x & 31;
    int half = lane >> 4;
    int l15  = lane & 15;
    int n0   = wave * 16;

    const __bf16* f0p = hbf  + (size_t)(n0 + l15) * EMB;
    const float*  f1p = aggr + (size_t)(n0 + l15) * EMB;

    v8f C[8];
#pragma unroll
    for (int nt = 0; nt < 8; ++nt) C[nt] = (v8f){0, 0, 0, 0, 0, 0, 0, 0};

#pragma unroll
    for (int kc = 0; kc < 8; ++kc) {
        int kb = kc * 32 + half * 8;
        v16bf a;
        if (kb < 128) {                          // bf16 activation half
            v8bf a0 = *(const v8bf*)(f0p + kb);
            v8bf a1 = *(const v8bf*)(f0p + kb + 16);
            a = __builtin_shufflevector(a0, a1, 0, 1, 2, 3, 4, 5, 6, 7,
                                        8, 9, 10, 11, 12, 13, 14, 15);
        } else {                                 // fp32 aggregate half -> cvt
            const float* g0p = f1p + (kb - 128);
#pragma unroll
            for (int i = 0; i < 8; ++i) a[i]     = (__bf16)g0p[i];
#pragma unroll
            for (int i = 0; i < 8; ++i) a[8 + i] = (__bf16)g0p[16 + i];
        }
#pragma unroll
        for (int nt = 0; nt < 8; ++nt) {
            const v8bf* wp = (const v8bf*)(sW + (((unsigned)(nt * 16 + l15)) << 8) + kb);
            v8bf b0 = wp[0];
            v8bf b1 = wp[2];
            v16bf b = __builtin_shufflevector(b0, b1, 0, 1, 2, 3, 4, 5, 6, 7,
                                              8, 9, 10, 11, 12, 13, 14, 15);
            C[nt] = __builtin_amdgcn_wmma_f32_16x16x32_bf16(
                false, a, false, b, (short)0, C[nt], false, false);
        }
    }

#pragma unroll
    for (int nt = 0; nt < 8; ++nt) {
        int col = nt * 16 + l15;
        float bi = bu[col];
#pragma unroll
        for (int r = 0; r < 8; ++r) {
            float v = C[nt][r] + bi;
            htmp[(size_t)(n0 + r + half * 8) * EMB + col] = v > 0.f ? v : 0.f;
        }
    }
}

__global__ void posupd_kernel(float* __restrict__ pos, const float* __restrict__ pos_aggr,
                              const float* __restrict__ cnt) {
    int t = blockIdx.x * blockDim.x + threadIdx.x;
    if (t >= NT) return;
    float c = fmaxf(cnt[t], 1.0f);
    pos[t * 2]     += pos_aggr[t * 2]     / c;
    pos[t * 2 + 1] += pos_aggr[t * 2 + 1] / c;
}

// ---------------------------------------------------------------------------
// PairNorm: column sums -> mean; global sum of squares -> scale; normalize
// and round once per layer into the bf16 activation buffer.
// ---------------------------------------------------------------------------
__global__ void colsum_kernel(const float* __restrict__ x, float* __restrict__ mean) {
    int t = blockIdx.x * blockDim.x + threadIdx.x;
    int c = t & 127, row = t >> 7;
    int rstride = (gridDim.x * blockDim.x) >> 7;
    float s = 0.f;
    for (int n = row; n < NT; n += rstride) s += x[(size_t)n * EMB + c];
    atomicAdd(&mean[c], s);
}
__global__ void sqsum_kernel(const float* __restrict__ x, const float* __restrict__ mean,
                             float* __restrict__ ss) {
    __shared__ float red[256];
    int t = blockIdx.x * blockDim.x + threadIdx.x;
    int c = t & 127, row = t >> 7;
    int rstride = (gridDim.x * blockDim.x) >> 7;
    float mu = mean[c] * (1.0f / NT);
    float s = 0.f;
    for (int n = row; n < NT; n += rstride) {
        float d = x[(size_t)n * EMB + c] - mu;
        s += d * d;
    }
    red[threadIdx.x] = s;
    __syncthreads();
    for (int off = 128; off > 0; off >>= 1) {
        if (threadIdx.x < off) red[threadIdx.x] += red[threadIdx.x + off];
        __syncthreads();
    }
    if (threadIdx.x == 0) atomicAdd(ss, red[0]);
}
__global__ void norm_kernel(const float* __restrict__ x, const float* __restrict__ mean,
                            const float* __restrict__ ss, __bf16* __restrict__ hbf) {
    int t = blockIdx.x * blockDim.x + threadIdx.x;
    if (t >= NT * EMB) return;
    int c = t & 127;
    float mu = mean[c] * (1.0f / NT);
    float sc = rsqrtf(ss[0] * (1.0f / NT) + 1e-5f);
    hbf[t] = (__bf16)((x[t] - mu) * sc);
}

// ---------------------------------------------------------------------------
// Readout: per-graph max pool + 2-layer head.
// ---------------------------------------------------------------------------
__global__ void pool_kernel(const __bf16* __restrict__ hbf, float* __restrict__ hg) {
    int t = blockIdx.x * blockDim.x + threadIdx.x;
    if (t >= BGR * EMB) return;
    int g = t >> 7, c = t & 127;
    float m = -3.0e38f;
    for (int n = 0; n < NNODE; ++n)
        m = fmaxf(m, (float)hbf[(size_t)(g * NNODE + n) * EMB + c]);
    hg[t] = m;
}
__global__ void head1_kernel(const float* __restrict__ hg, const float* __restrict__ W1,
                             const float* __restrict__ b1, float* __restrict__ t1) {
    int t = blockIdx.x * blockDim.x + threadIdx.x;
    if (t >= BGR * EMB) return;
    int g = t >> 7, c = t & 127;
    float s = b1[c];
    for (int k = 0; k < EMB; ++k) s += hg[g * EMB + k] * W1[k * EMB + c];
    t1[t] = s > 0.f ? s : 0.f;
}
__global__ void head2_kernel(const float* __restrict__ t1, const float* __restrict__ W2,
                             const float* __restrict__ b2, float* __restrict__ out) {
    int t = blockIdx.x * blockDim.x + threadIdx.x;
    if (t >= BGR * 2) return;
    int g = t >> 1, c = t & 1;
    float s = b2[c];
    for (int k = 0; k < EMB; ++k) s += t1[g * EMB + k] * W2[k * 2 + c];
    out[t] = s;
}

// ---------------------------------------------------------------------------
extern "C" void kernel_launch(void* const* d_in, const int* in_sizes, int n_in,
                              void* d_out, int out_size, void* d_ws, size_t ws_size,
                              hipStream_t stream) {
    const float* h_feat = (const float*)d_in[0];
    const float* pos_in = (const float*)d_in[1];
    const int*   ei     = (const int*)  d_in[2];
    const float* W_emb  = (const float*)d_in[3];
    const float* b_emb  = (const float*)d_in[4];
    const float* W_msg  = (const float*)d_in[5];
    const float* b_msg  = (const float*)d_in[6];
    const float* W_pos  = (const float*)d_in[7];
    const float* b_pos  = (const float*)d_in[8];
    const float* W_upd  = (const float*)d_in[9];
    const float* b_upd  = (const float*)d_in[10];
    const float* W1     = (const float*)d_in[11];
    const float* b1     = (const float*)d_in[12];
    const float* W2     = (const float*)d_in[13];
    const float* b2     = (const float*)d_in[14];
    float* out = (float*)d_out;

    // Workspace carve-up (256B aligned)
    char* w = (char*)d_ws;
    auto alloc = [&](size_t bytes) -> void* {
        void* p = (void*)w;
        w += (bytes + 255) & ~(size_t)255;
        return p;
    };
    __bf16* hbf      = (__bf16*)alloc((size_t)NT * EMB * 2);  // bf16 activations
    float*  htmp     = (float*)alloc((size_t)NT * EMB * 4);
    float*  msg_aggr = (float*)alloc((size_t)NT * EMB * 4);   // zero block start
    float*  pos_aggr = (float*)alloc((size_t)NT * 2 * 4);
    float*  cnt      = (float*)alloc((size_t)NT * 4);
    float*  mean     = (float*)alloc(EMB * 4);
    float*  ss       = (float*)alloc(256);                    // zero block end
    char*   zend     = w;
    float*  pos      = (float*)alloc((size_t)NT * 2 * 4);
    int*    src_ws   = (int*)  alloc((size_t)NE * 4);
    int*    dst_ws   = (int*)  alloc((size_t)NE * 4);
    __bf16* WmT      = (__bf16*)alloc((size_t)LAY * EMB * 257 * 2);
    __bf16* WuT      = (__bf16*)alloc((size_t)LAY * EMB * 256 * 2);
    float*  hg       = (float*)alloc(BGR * EMB * 4);
    float*  t1       = (float*)alloc(BGR * EMB * 4);

    const int zspanf = (int)((zend - (char*)msg_aggr) / 4);

    // Weight prep (bf16 transposed)
    {
        int n1 = LAY * 257 * EMB;
        cvt_wmsg_kernel<<<(n1 + 255) / 256, 256, 0, stream>>>(W_msg, WmT);
        int n2 = LAY * 256 * EMB;
        cvt_wupd_kernel<<<(n2 + 255) / 256, 256, 0, stream>>>(W_upd, WuT);
    }

    // Embed + copy pos
    embed_kernel<<<(NT * EMB + 255) / 256, 256, 0, stream>>>(h_feat, W_emb, b_emb, hbf,
                                                             pos_in, pos);

    const int etiles = NE / 16;   // 40000 -> 5000 blocks of 8 waves (exact)
    const int ntiles = NT / 16;   // 4000  -> 500 blocks of 8 waves (exact)

    for (int l = 0; l < LAY; ++l) {
        const int* src = (l == 0) ? ei      : src_ws;
        const int* dst = (l == 0) ? ei + NE : dst_ws;
        if (l > 0) {
            dim3 g((NNODE + 255) / 256, BGR);
            knn_kernel<<<g, 256, 0, stream>>>(pos, src_ws, dst_ws);
        }
        zero_kernel<<<(zspanf + 255) / 256, 256, 0, stream>>>(msg_aggr, zspanf);
        cnt_kernel<<<(NE + 255) / 256, 256, 0, stream>>>(dst, cnt);

        msg_kernel<<<etiles * 32 / 256, 256, 0, stream>>>(
            hbf, pos, src, dst, WmT + (size_t)l * EMB * 257, b_msg + l * EMB,
            W_pos + l * EMB * 2, b_pos + l * 2, msg_aggr, pos_aggr);

        upd_kernel<<<ntiles * 32 / 256, 256, 0, stream>>>(
            hbf, msg_aggr, WuT + (size_t)l * EMB * 256, b_upd + l * EMB, htmp);

        posupd_kernel<<<(NT + 255) / 256, 256, 0, stream>>>(pos, pos_aggr, cnt);

        // PairNorm: htmp -> hbf (round once per layer into bf16 activations)
        colsum_kernel<<<500, 256, 0, stream>>>(htmp, mean);
        sqsum_kernel<<<500, 256, 0, stream>>>(htmp, mean, ss);
        norm_kernel<<<(NT * EMB + 255) / 256, 256, 0, stream>>>(htmp, mean, ss, hbf);
    }

    // Readout
    pool_kernel<<<(BGR * EMB + 255) / 256, 256, 0, stream>>>(hbf, hg);
    head1_kernel<<<(BGR * EMB + 255) / 256, 256, 0, stream>>>(hg, W1, b1, t1);
    head2_kernel<<<1, 256, 0, stream>>>(t1, W2, b2, out);
}
